// FeatureProcessor_85779086836067
// MI455X (gfx1250) — compile-verified, matrix operand order checked by reading, not run on hardware
//
#include <hip/hip_runtime.h>
#include <hip/hip_bf16.h>

typedef __attribute__((ext_vector_type(16))) _Float16 v16h;
typedef __attribute__((ext_vector_type(8)))  float    v8f;

#define BATCH  8
#define HDIM   64
#define WDIM   64
#define HW     4096     // 64*64
#define BHW    32768    // 8*64*64
#define LOG_HW 12
#define LOG_W  6

__device__ __forceinline__ float rdscale(const float* p) { return p ? p[0] : 1.0f; }

// ---------------------------------------------------------------------------
// Weight packing: OIHW f32 -> f16 A-fragment layout
// packed[ocTile][kChunk][lane][16], lane: m = lane&15, half = lane>>4,
// slot 2j+e -> k = (j<4 ? 2j+8*half : 16+2*(j-4)+8*half) + e  (ISA 16-bit A 16x32)
// ---------------------------------------------------------------------------
__global__ __launch_bounds__(256) void pack_weights_kernel(
    const float* __restrict__ w, _Float16* __restrict__ packed,
    int OC, int K, int ocTiles, int kChunks)
{
  size_t total = (size_t)ocTiles * kChunks * 512;
  for (size_t i = (size_t)blockIdx.x * blockDim.x + threadIdx.x; i < total;
       i += (size_t)gridDim.x * blockDim.x) {
    int slot = (int)(i & 511);
    size_t rest = i >> 9;
    int chunk = (int)(rest % kChunks);
    int ocTile = (int)(rest / kChunks);
    int lane = slot >> 4;
    int s = slot & 15;
    int j = s >> 1;
    int e = s & 1;
    int half = lane >> 4;
    int m = lane & 15;
    int k = (j < 4 ? 2 * j + 8 * half : 16 + 2 * (j - 4) + 8 * half) + e;
    int kg = chunk * 32 + k;
    int oc = ocTile * 16 + m;
    float v = (oc < OC && kg < K) ? w[(size_t)oc * K + kg] : 0.0f;
    packed[i] = (_Float16)v;
  }
}

// k-decode table: kg -> (ic<<16)|((kh*dil-pad+64)<<8)|(kw*dil-pad+64)
__global__ __launch_bounds__(256) void decode_table_kernel(
    int* __restrict__ tbl, int K, int KHW, int KW, int pad, int dil)
{
  for (int kg = blockIdx.x * blockDim.x + threadIdx.x; kg < K;
       kg += gridDim.x * blockDim.x) {
    int ic = kg / KHW;
    int r  = kg - ic * KHW;
    int kh = r / KW;
    int kw = r - kh * KW;
    int dy = kh * dil - pad + 64;
    int dx = kw * dil - pad + 64;
    tbl[kg] = (ic << 16) | (dy << 8) | dx;
  }
}

// ---------------------------------------------------------------------------
// Implicit-GEMM convolution, WMMA f32_16x16x32_f16.
// Block = 128 threads (4 waves). Block tile: 128 oc x 64 pixels (one full
// image row => fixed (b,oh), coalesced row reads). Each wave: 2 oc-tiles x
// 4 N-subtiles = 8 accumulators, reusing B fragments across both A tiles
// (8 wmma per chunk per wave).
// Staging: thread t owns K-row k=t>>2, pixel segment seg=t&3 (16 pixels):
// one k-table lookup + one row base per chunk, inner loop is load/cvt/ds.
// act: 0=none 1=relu 2=leaky(0.1) 3=tanh
// ---------------------------------------------------------------------------
__global__ __launch_bounds__(128) void conv_wmma_kernel(
    const float* __restrict__ src0, int C0full, int off0, int n0, const float* __restrict__ sc0,
    const float* __restrict__ src1, int C1full, int off1, const float* __restrict__ sc1,
    const int* __restrict__ ktab,
    const _Float16* __restrict__ packA, const float* __restrict__ bias,
    float* __restrict__ out, int OC, int Ktot, int kChunks, int act)
{
  __shared__ alignas(64) _Float16 ldsB[2048];   // 32(K) x 64(N) f16, 4 B-fragments
  const int tid  = threadIdx.x;
  const int lane = tid & 31;
  const int wave = tid >> 5;
  const int b    = blockIdx.x >> 6;     // row tile: fixed batch + output row
  const int oh   = blockIdx.x & 63;
  const int ocTile0 = (blockIdx.y * 4 + wave) * 2;
  const bool act0 = (ocTile0 * 16) < OC;
  const bool act1 = ((ocTile0 + 1) * 16) < OC;
  const float s0 = rdscale(sc0);
  const float s1 = rdscale(sc1);

  // staging role for this thread: one K-row, one 16-pixel segment
  const int km  = tid >> 2;             // 0..31  k within chunk
  const int seg = tid & 3;              // 0..3   N-subtile
  // fragment-swizzle constants for this k (hoisted out of the pixel loop)
  {
  }
  const int hi   = km >> 4;
  const int r4   = km & 15;
  const int halfk = r4 >> 3;
  const int jk   = hi * 4 + ((r4 & 7) >> 1);
  const int ek   = r4 & 1;
  _Float16* ldsSlot = &ldsB[seg * 512 + (halfk * 16) * 16 + 2 * jk + ek];

  v8f acc00 = {}, acc01 = {}, acc02 = {}, acc03 = {};
  v8f acc10 = {}, acc11 = {}, acc12 = {}, acc13 = {};

  for (int chunk = 0; chunk < kChunks; ++chunk) {
    __syncthreads();  // protect previous iteration's LDS reads
    // ---- stage 32K x 64N activation tile (16 pixels per thread) ----
    {
      int kg = chunk * 32 + km;
      const float* rowp = nullptr;
      float scl = 1.0f;
      int dx = 0;
      if (kg < Ktot) {
        int t3 = ktab[kg];
        int ic = t3 >> 16;
        int ih = oh + ((t3 >> 8) & 255) - 64;
        dx = (t3 & 255) - 64;
        if (ih >= 0 && ih < HDIM) {
          if (ic < n0) {
            rowp = src0 + (((size_t)(b * C0full + off0 + ic)) << LOG_HW) + (ih << LOG_W);
            scl = s0;
          } else {
            rowp = src1 + (((size_t)(b * C1full + off1 + ic - n0)) << LOG_HW) + (ih << LOG_W);
            scl = s1;
          }
        }
      }
      const int iwBase = seg * 16 + dx;
#pragma unroll
      for (int i = 0; i < 16; ++i) {
        int iw = iwBase + i;
        float v = (rowp && iw >= 0 && iw < WDIM) ? rowp[iw] * scl : 0.0f;
        ldsSlot[i * 16] = (_Float16)v;
      }
    }
    __syncthreads();
    // ---- compute: 2 A fragments x 4 B fragments ----
    if (act0 || act1) {
      v16h b0 = *(const v16h*)&ldsB[0    + lane * 16];
      v16h b1 = *(const v16h*)&ldsB[512  + lane * 16];
      v16h b2 = *(const v16h*)&ldsB[1024 + lane * 16];
      v16h b3 = *(const v16h*)&ldsB[1536 + lane * 16];
      const v16h* ap0 = (const v16h*)packA + ((size_t)ocTile0 * kChunks + chunk) * 32 + lane;
      if (chunk + 1 < kChunks)
        __builtin_prefetch((const void*)(ap0 + 32), 0, 1);   // next chunk A stream
      if (act0) {
        v16h a0 = *ap0;
        acc00 = __builtin_amdgcn_wmma_f32_16x16x32_f16(false, a0, false, b0, (short)0, acc00, false, false);
        acc01 = __builtin_amdgcn_wmma_f32_16x16x32_f16(false, a0, false, b1, (short)0, acc01, false, false);
        acc02 = __builtin_amdgcn_wmma_f32_16x16x32_f16(false, a0, false, b2, (short)0, acc02, false, false);
        acc03 = __builtin_amdgcn_wmma_f32_16x16x32_f16(false, a0, false, b3, (short)0, acc03, false, false);
      }
      if (act1) {
        v16h a1 = *((const v16h*)packA + ((size_t)(ocTile0 + 1) * kChunks + chunk) * 32 + lane);
        acc10 = __builtin_amdgcn_wmma_f32_16x16x32_f16(false, a1, false, b0, (short)0, acc10, false, false);
        acc11 = __builtin_amdgcn_wmma_f32_16x16x32_f16(false, a1, false, b1, (short)0, acc11, false, false);
        acc12 = __builtin_amdgcn_wmma_f32_16x16x32_f16(false, a1, false, b2, (short)0, acc12, false, false);
        acc13 = __builtin_amdgcn_wmma_f32_16x16x32_f16(false, a1, false, b3, (short)0, acc13, false, false);
      }
    }
  }
  // ---- epilogue: bias + activation, scatter to NCHW ----
  const int halfd = lane >> 4;
  const int n = lane & 15;
  const v8f accA[2][4] = {{acc00, acc01, acc02, acc03}, {acc10, acc11, acc12, acc13}};
#pragma unroll
  for (int mt = 0; mt < 2; ++mt) {
    int ocT = ocTile0 + mt;
    if (ocT * 16 < OC) {
#pragma unroll
      for (int t = 0; t < 4; ++t) {
        int hw = (oh << LOG_W) + t * 16 + n;
#pragma unroll
        for (int r = 0; r < 8; ++r) {
          int m = r + 8 * halfd;
          int oc = ocT * 16 + m;
          float v = accA[mt][t][r] + bias[oc];
          if (act == 1)      v = fmaxf(v, 0.0f);
          else if (act == 2) v = (v > 0.0f) ? v : 0.1f * v;
          else if (act == 3) v = tanhf(v);
          out[(((size_t)(b * OC + oc)) << LOG_HW) + hw] = v;
        }
      }
    }
  }
}

// ---------------------------------------------------------------------------
// BatchNorm (training-mode batch stats over N,H,W), stats[2c]=mean, [2c+1]=var
// ---------------------------------------------------------------------------
__global__ __launch_bounds__(256) void bn_stats_kernel(
    const float* __restrict__ x, float* __restrict__ stats, int C)
{
  __shared__ float sm[256], sm2[256];
  int c = blockIdx.x;
  float s = 0.0f, s2 = 0.0f;
  for (int i = threadIdx.x; i < BHW; i += 256) {
    int b = i >> LOG_HW;
    int hw = i & (HW - 1);
    float v = x[(((size_t)(b * C + c)) << LOG_HW) + hw];
    s += v; s2 += v * v;
  }
  sm[threadIdx.x] = s; sm2[threadIdx.x] = s2;
  __syncthreads();
  for (int o = 128; o > 0; o >>= 1) {
    if (threadIdx.x < o) {
      sm[threadIdx.x]  += sm[threadIdx.x + o];
      sm2[threadIdx.x] += sm2[threadIdx.x + o];
    }
    __syncthreads();
  }
  if (threadIdx.x == 0) {
    float m = sm[0] * (1.0f / (float)BHW);
    stats[2 * c]     = m;
    stats[2 * c + 1] = sm2[0] * (1.0f / (float)BHW) - m * m;
  }
}

__global__ __launch_bounds__(256) void bn_apply_kernel(
    float* __restrict__ x, const float* __restrict__ stats,
    const float* __restrict__ g, const float* __restrict__ bb, int C, int relu)
{
  size_t N = (size_t)BATCH * C * HW;
  for (size_t i = (size_t)blockIdx.x * blockDim.x + threadIdx.x; i < N;
       i += (size_t)gridDim.x * blockDim.x) {
    int c = (int)((i >> LOG_HW) % C);
    float m = stats[2 * c], v = stats[2 * c + 1];
    float y = (x[i] - m) * rsqrtf(v + 1e-5f) * g[c] + bb[c];
    if (relu) y = fmaxf(y, 0.0f);
    x[i] = y;
  }
}

// ---------------------------------------------------------------------------
// 5x5 reflect-padded box sum (scaled by w_hh)
// ---------------------------------------------------------------------------
__device__ __forceinline__ int refl(int t) { return t < 0 ? -t : (t > 63 ? 126 - t : t); }

__global__ __launch_bounds__(256) void boxsum5_kernel(
    const float* __restrict__ hh, const float* __restrict__ whh, float* __restrict__ out)
{
  const float s = rdscale(whh);
  size_t N = (size_t)BATCH * 64 * HW;
  for (size_t i = (size_t)blockIdx.x * blockDim.x + threadIdx.x; i < N;
       i += (size_t)gridDim.x * blockDim.x) {
    size_t bc = i >> LOG_HW;
    int hw = (int)(i & (HW - 1));
    int oh = hw >> LOG_W, ow = hw & 63;
    float acc = 0.0f;
    for (int dy = -2; dy <= 2; ++dy) {
      int ih = refl(oh + dy);
      for (int dx = -2; dx <= 2; ++dx) {
        int iw = refl(ow + dx);
        acc += hh[(bc << LOG_HW) + (ih << LOG_W) + iw];
      }
    }
    out[i] = acc * s;
  }
}

// median over 64 channels per pixel (even count -> mean of two mid stats)
__global__ __launch_bounds__(256) void median64_kernel(
    const float* __restrict__ s, float* __restrict__ med)
{
  int p = blockIdx.x * blockDim.x + threadIdx.x;
  if (p >= BHW) return;
  int b = p >> LOG_HW;
  int hw = p & (HW - 1);
  float v[64];
  for (int c = 0; c < 64; ++c) {
    float x = s[(((size_t)(b * 64 + c)) << LOG_HW) + hw];
    int j = c;
    while (j > 0 && v[j - 1] > x) { v[j] = v[j - 1]; --j; }
    v[j] = x;
  }
  med[p] = 0.5f * (v[31] + v[32]);
}

// hh_l1 = broadcast(med) + hh_freq
__global__ __launch_bounds__(256) void addmed_kernel(
    const float* __restrict__ med, const float* __restrict__ hh, float* __restrict__ out)
{
  size_t N = (size_t)BATCH * 64 * HW;
  for (size_t i = (size_t)blockIdx.x * blockDim.x + threadIdx.x; i < N;
       i += (size_t)gridDim.x * blockDim.x) {
    size_t bc = i >> LOG_HW;
    int b = (int)(bc >> 6);
    int hw = (int)(i & (HW - 1));
    out[i] = med[((size_t)b << LOG_HW) + hw] + hh[i];
  }
}

// x = concat(w_hl*hl, w_lh*lh) * sigmoid(x), in place on (B,128,H,W)
__global__ __launch_bounds__(256) void gate_kernel(
    float* __restrict__ x, const float* __restrict__ hl, const float* __restrict__ lh,
    const float* __restrict__ whl, const float* __restrict__ wlh)
{
  const float s0 = rdscale(whl), s1 = rdscale(wlh);
  size_t N = (size_t)BATCH * 128 * HW;
  for (size_t i = (size_t)blockIdx.x * blockDim.x + threadIdx.x; i < N;
       i += (size_t)gridDim.x * blockDim.x) {
    size_t bc = i >> LOG_HW;
    int c = (int)(bc & 127);
    int b = (int)(bc >> 7);
    int hw = (int)(i & (HW - 1));
    float orig = (c < 64)
        ? s0 * hl[(((size_t)(b * 64 + c))      << LOG_HW) + hw]
        : s1 * lh[(((size_t)(b * 64 + c - 64)) << LOG_HW) + hw];
    float g = 1.0f / (1.0f + expf(-x[i]));
    x[i] = orig * g;
  }
}

// Haar IDWT (pywt idwt2 convention) -> (B,C,2H,2W)
__global__ __launch_bounds__(256) void idwt_kernel(
    const float* __restrict__ ca, const float* __restrict__ ch,
    const float* __restrict__ cv, const float* __restrict__ cd,
    float* __restrict__ out)
{
  size_t N = (size_t)BATCH * 64 * 128 * 128;
  for (size_t i = (size_t)blockIdx.x * blockDim.x + threadIdx.x; i < N;
       i += (size_t)gridDim.x * blockDim.x) {
    int ow2 = (int)(i & 127);
    int oh2 = (int)((i >> 7) & 127);
    size_t bc = i >> 14;
    int h = oh2 >> 1, w = ow2 >> 1;
    int r = oh2 & 1, s = ow2 & 1;
    size_t idx = (bc << LOG_HW) + (h << LOG_W) + w;
    float A = ca[idx], Hc = ch[idx], V = cv[idx], D = cd[idx];
    float v;
    if (!r && !s)      v = A + Hc + V + D;
    else if (!r && s)  v = A + Hc - V - D;
    else if (r && !s)  v = A - Hc + V - D;
    else               v = A - Hc - V + D;
    out[i] = 0.5f * v;
  }
}

// ---------------------------------------------------------------------------
// Host orchestration
// ---------------------------------------------------------------------------
static void launch_conv(hipStream_t st,
                        const float* s0, int C0f, int o0, int n0, const float* sc0,
                        const float* s1, int C1f, int o1, const float* sc1,
                        const int* ktab, const _Float16* pA, const float* bias,
                        float* out, int OC, int IC, int KH, int KW, int act)
{
  int K = IC * KH * KW;
  int kChunks = (K + 31) / 32;
  dim3 grid(BHW / 64, (OC + 127) / 128);
  conv_wmma_kernel<<<grid, 128, 0, st>>>(s0, C0f, o0, n0, sc0, s1, C1f, o1, sc1,
                                         ktab, pA, bias, out, OC, K, kChunks, act);
}

static void launch_pack(hipStream_t st, const float* w, _Float16* packed, int OC, int K)
{
  int ocTiles = (OC + 15) / 16;
  int kChunks = (K + 31) / 32;
  pack_weights_kernel<<<2048, 256, 0, st>>>(w, packed, OC, K, ocTiles, kChunks);
}

static void launch_table(hipStream_t st, int* tbl, int IC, int KH, int KW, int pad, int dil)
{
  int K = IC * KH * KW;
  decode_table_kernel<<<64, 256, 0, st>>>(tbl, K, KH * KW, KW, pad, dil);
}

static size_t packed_bytes(int OC, int K)
{
  return (size_t)((OC + 15) / 16) * ((K + 31) / 32) * 512 * 2;
}

extern "C" void kernel_launch(void* const* d_in, const int* in_sizes, int n_in,
                              void* d_out, int out_size, void* d_ws, size_t ws_size,
                              hipStream_t stream)
{
  (void)in_sizes; (void)n_in; (void)out_size; (void)ws_size;
  const float* ll_f  = (const float*)d_in[0];
  const float* hl_f  = (const float*)d_in[1];
  const float* lh_f  = (const float*)d_in[2];
  const float* hh_f  = (const float*)d_in[3];
  const float* w_ll  = (const float*)d_in[4];
  const float* w_hl  = (const float*)d_in[5];
  const float* w_lh  = (const float*)d_in[6];
  const float* w_hh  = (const float*)d_in[7];
  const float* cc_w  = (const float*)d_in[8];   const float* cc_b  = (const float*)d_in[9];
  const float* bnc_g = (const float*)d_in[10];  const float* bnc_b = (const float*)d_in[11];
  const float* e1_w  = (const float*)d_in[12];  const float* e1_b  = (const float*)d_in[13];
  const float* fe_w  = (const float*)d_in[14];  const float* fe_b  = (const float*)d_in[15];
  const float* e2_w  = (const float*)d_in[16];  const float* e2_b  = (const float*)d_in[17];
  const float* bne_g = (const float*)d_in[18];  const float* bne_b = (const float*)d_in[19];
  const float* cr_w  = (const float*)d_in[20];  const float* cr_b  = (const float*)d_in[21];
  const float* bnr_g = (const float*)d_in[22];  const float* bnr_b = (const float*)d_in[23];
  const float* adj_w = (const float*)d_in[24];  const float* adj_b = (const float*)d_in[25];
  const float* cll_w = (const float*)d_in[26];  const float* cll_b = (const float*)d_in[27];
  const float* chh_w = (const float*)d_in[28];  const float* chh_b = (const float*)d_in[29];
  const float* c1_w  = (const float*)d_in[30];  const float* c1_b  = (const float*)d_in[31];
  const float* c3_w  = (const float*)d_in[32];  const float* c3_b  = (const float*)d_in[33];
  const float* c5_w  = (const float*)d_in[34];  const float* c5_b  = (const float*)d_in[35];

  // --- workspace bump allocator ---
  char* base = (char*)d_ws;
  size_t off = 0;
  auto alloc = [&](size_t bytes) -> void* {
    void* p = base + off;
    off = (off + bytes + 255) & ~(size_t)255;
    return p;
  };
  const size_t S    = (size_t)BATCH * 64  * HW * 4;   // 8 MB
  const size_t S16  = (size_t)BATCH * 16  * HW * 4;
  const size_t S128 = (size_t)BATCH * 128 * HW * 4;
  const size_t S512 = (size_t)BATCH * 512 * HW * 4;

  float* bsum  = (float*)alloc(S);
  float* med   = (float*)alloc((size_t)BHW * 4);
  float* hh_l1 = (float*)alloc(S);
  float* t16a  = (float*)alloc(S16);
  float* t16b  = (float*)alloc(S16);
  float* t512a = (float*)alloc(S512);
  float* t512b = (float*)alloc(S512);
  float* t128  = (float*)alloc(S128);
  float* la    = (float*)alloc(S);
  float* lb    = (float*)alloc(S);
  float* ll2   = (float*)alloc(S);
  float* hl2   = (float*)alloc(S);
  float* lh2   = (float*)alloc(S);
  float* hh2   = (float*)alloc(S);
  float* stats = (float*)alloc(1024 * 4);

  _Float16* p_cc  = (_Float16*)alloc(packed_bytes(16, 128));
  _Float16* p_e1  = (_Float16*)alloc(packed_bytes(16, 144));
  _Float16* p_fe  = (_Float16*)alloc(packed_bytes(512, 16));
  _Float16* p_e2  = (_Float16*)alloc(packed_bytes(512, 12800));
  _Float16* p_cr  = (_Float16*)alloc(packed_bytes(128, 512));
  _Float16* p_adj = (_Float16*)alloc(packed_bytes(64, 128));
  _Float16* p_cll = (_Float16*)alloc(packed_bytes(64, 1600));
  _Float16* p_chh = (_Float16*)alloc(packed_bytes(64, 64));
  _Float16* p_c1  = (_Float16*)alloc(packed_bytes(64, 576));
  _Float16* p_c3  = (_Float16*)alloc(packed_bytes(64, 1600));
  _Float16* p_c5  = (_Float16*)alloc(packed_bytes(64, 3136));

  int* tb_cc  = (int*)alloc(128   * 4);
  int* tb_e1  = (int*)alloc(144   * 4);
  int* tb_fe  = (int*)alloc(16    * 4);
  int* tb_e2  = (int*)alloc(12800 * 4);
  int* tb_cr  = (int*)alloc(512   * 4);
  int* tb_adj = (int*)alloc(128   * 4);
  int* tb_cll = (int*)alloc(1600  * 4);
  int* tb_chh = (int*)alloc(64    * 4);
  int* tb_c1  = (int*)alloc(576   * 4);
  int* tb_c3  = (int*)alloc(1600  * 4);
  int* tb_c5  = (int*)alloc(3136  * 4);

  // --- pack weights + build k-decode tables ---
  launch_pack(stream, cc_w,  p_cc,  16,  128);
  launch_pack(stream, e1_w,  p_e1,  16,  144);
  launch_pack(stream, fe_w,  p_fe,  512, 16);
  launch_pack(stream, e2_w,  p_e2,  512, 12800);
  launch_pack(stream, cr_w,  p_cr,  128, 512);
  launch_pack(stream, adj_w, p_adj, 64,  128);
  launch_pack(stream, cll_w, p_cll, 64,  1600);
  launch_pack(stream, chh_w, p_chh, 64,  64);
  launch_pack(stream, c1_w,  p_c1,  64,  576);
  launch_pack(stream, c3_w,  p_c3,  64,  1600);
  launch_pack(stream, c5_w,  p_c5,  64,  3136);

  launch_table(stream, tb_cc,  128, 1, 1, 0, 1);
  launch_table(stream, tb_e1,  16,  3, 3, 1, 1);
  launch_table(stream, tb_fe,  16,  1, 1, 0, 1);
  launch_table(stream, tb_e2,  512, 5, 5, 2, 1);
  launch_table(stream, tb_cr,  512, 1, 1, 0, 1);
  launch_table(stream, tb_adj, 128, 1, 1, 0, 1);
  launch_table(stream, tb_cll, 64,  5, 5, 4, 2);
  launch_table(stream, tb_chh, 64,  1, 1, 0, 1);
  launch_table(stream, tb_c1,  64,  3, 3, 1, 1);
  launch_table(stream, tb_c3,  64,  5, 5, 2, 1);
  launch_table(stream, tb_c5,  64,  7, 7, 3, 1);

  // --- HH branch: median of 5x5 box sums over channels ---
  boxsum5_kernel<<<2048, 256, 0, stream>>>(hh_f, w_hh, bsum);
  median64_kernel<<<(BHW + 255) / 256, 256, 0, stream>>>(bsum, med);
  addmed_kernel<<<2048, 256, 0, stream>>>(med, hh_f, hh_l1);

  // --- CCFEA SE block on concat(w_hl*hl, w_lh*lh) ---
  launch_conv(stream, hl_f, 64, 0, 64, w_hl, lh_f, 64, 0, w_lh,
              tb_cc, p_cc, cc_b, t16a, 16, 128, 1, 1, 0);
  launch_conv(stream, t16a, 16, 0, 16, nullptr, nullptr, 0, 0, nullptr,
              tb_e1, p_e1, e1_b, t16b, 16, 16, 3, 3, 0);
  bn_stats_kernel<<<16, 256, 0, stream>>>(t16b, stats, 16);
  bn_apply_kernel<<<2048, 256, 0, stream>>>(t16b, stats, bnc_g, bnc_b, 16, 1);
  launch_conv(stream, t16b, 16, 0, 16, nullptr, nullptr, 0, 0, nullptr,
              tb_fe, p_fe, fe_b, t512a, 512, 16, 1, 1, 0);
  launch_conv(stream, t512a, 512, 0, 512, nullptr, nullptr, 0, 0, nullptr,
              tb_e2, p_e2, e2_b, t512b, 512, 512, 5, 5, 0);
  bn_stats_kernel<<<512, 256, 0, stream>>>(t512b, stats, 512);
  bn_apply_kernel<<<4096, 256, 0, stream>>>(t512b, stats, bne_g, bne_b, 512, 1);
  launch_conv(stream, t512b, 512, 0, 512, nullptr, nullptr, 0, 0, nullptr,
              tb_cr, p_cr, cr_b, t128, 128, 512, 1, 1, 0);
  bn_stats_kernel<<<128, 256, 0, stream>>>(t128, stats, 128);
  bn_apply_kernel<<<2048, 256, 0, stream>>>(t128, stats, bnr_g, bnr_b, 128, 0);
  gate_kernel<<<2048, 256, 0, stream>>>(t128, hl_f, lh_f, w_hl, w_lh);

  // --- LL multi-kernel branch (w_ll folded into first conv's staging) ---
  launch_conv(stream, ll_f, 64, 0, 64, w_ll, nullptr, 0, 0, nullptr,
              tb_c1, p_c1, c1_b, la, 64, 64, 3, 3, 0);
  launch_conv(stream, la, 64, 0, 64, nullptr, nullptr, 0, 0, nullptr,
              tb_c3, p_c3, c3_b, lb, 64, 64, 5, 5, 0);
  launch_conv(stream, lb, 64, 0, 64, nullptr, nullptr, 0, 0, nullptr,
              tb_c5, p_c5, c5_b, la, 64, 64, 7, 7, 0);

  // --- adj 1x1 convs on concat(branch, skip) ---
  launch_conv(stream, la, 64, 0, 64, nullptr, ll_f, 64, 0, w_ll,
              tb_adj, p_adj, adj_b, ll2, 64, 128, 1, 1, 0);
  launch_conv(stream, t128, 128, 0, 64, nullptr, hl_f, 64, 0, nullptr,
              tb_adj, p_adj, adj_b, hl2, 64, 128, 1, 1, 0);
  launch_conv(stream, t128, 128, 64, 64, nullptr, lh_f, 64, 0, nullptr,
              tb_adj, p_adj, adj_b, lh2, 64, 128, 1, 1, 0);

  // --- HH tail: 1x1 + leaky(0.1), then 5x5 dil=2 + tanh ---
  launch_conv(stream, hh_l1, 64, 0, 64, nullptr, nullptr, 0, 0, nullptr,
              tb_chh, p_chh, chh_b, lb, 64, 64, 1, 1, 2);
  launch_conv(stream, lb, 64, 0, 64, nullptr, nullptr, 0, 0, nullptr,
              tb_cll, p_cll, cll_b, hh2, 64, 64, 5, 5, 3);

  // --- Haar IDWT -> output (8,64,128,128) ---
  idwt_kernel<<<4096, 256, 0, stream>>>(ll2, hl2, lh2, hh2, (float*)d_out);
}
